// Attention_7464653161210
// MI455X (gfx1250) — matrix operand order repaired and checked by reading, not compiled, not run on hardware
//
#include <hip/hip_runtime.h>
#include <hip/hip_bf16.h>

typedef __attribute__((ext_vector_type(16))) _Float16 v16h;
typedef __attribute__((ext_vector_type(8)))  _Float16 v8h;
typedef __attribute__((ext_vector_type(8)))  float    v8f;

#define WMMA16(a, b, c) \
  __builtin_amdgcn_wmma_f32_16x16x32_f16(false, (a), false, (b), (short)0, (c), false, false)

constexpr int Bq  = 4;
constexpr int Nn  = 2048;
constexpr int Cc  = 128;
constexpr int Hh  = 4;
constexpr int HDd = 32;
constexpr float SCALE = 0.17677669529663687f;  // 1/sqrt(32)
constexpr float NEGV  = -9.0e15f;

// ---------------------------------------------------------------------------
// CDNA5 async global->LDS copy (ASYNCcnt-tracked). LDS dest address is the
// low 32 bits of the flat shared pointer.
// ---------------------------------------------------------------------------
__device__ __forceinline__ void async_ld_b128(void* lds, const void* gaddr) {
  unsigned ldsoff = (unsigned)(size_t)lds;
  asm volatile("global_load_async_to_lds_b128 %0, %1, off"
               :: "v"(ldsoff), "v"(gaddr)
               : "memory");
}
__device__ __forceinline__ void wait_async0() {
  asm volatile("s_wait_asynccnt 0" ::: "memory");
}

// ---------------------------------------------------------------------------
// Fragment loaders (wave32, ISA 7.12.2 layouts).
// A-matrix 16x32 f16: lanes 0-15 row=lane, elems[0..7]=K0..7, [8..15]=K16..23
//                     lanes 16-31 row=lane-16, [0..7]=K8..15, [8..15]=K24..31
// ---------------------------------------------------------------------------
__device__ __forceinline__ v16h load_fragA(const _Float16* base, int stride) {
  const int lane = threadIdx.x & 31;
  const int row  = lane & 15;
  const int k0   = (lane >> 4) << 3;  // 0 or 8
  const _Float16* p = base + row * stride + k0;
  v8h lo = *(const v8h*)(p);
  v8h hi = *(const v8h*)(p + 16);
  v16h f;
#pragma unroll
  for (int i = 0; i < 8; ++i) { f[i] = lo[i]; f[i + 8] = hi[i]; }
  return f;
}

// B-matrix 32x16 f16: lane holds one output column (=memory row of the
// row-major "other" operand); lanes 0-15 K=0..7,16..23, lanes 16-31 K=8..15,24..31.
__device__ __forceinline__ v16h load_fragB(const _Float16* base, int stride) {
  const int lane = threadIdx.x & 31;
  const int col  = lane & 15;
  const int k0   = (lane >> 4) << 3;  // 0 or 8
  const _Float16* p = base + col * stride + k0;
  v8h lo = *(const v8h*)(p);
  v8h hi = *(const v8h*)(p + 16);
  v16h f;
#pragma unroll
  for (int i = 0; i < 8; ++i) { f[i] = lo[i]; f[i + 8] = hi[i]; }
  return f;
}

// ---------------------------------------------------------------------------
// 1) fp32 -> f16 conversion of x and weights
// ---------------------------------------------------------------------------
__global__ __launch_bounds__(256) void cvt_kernel(
    const float* __restrict x, const float* __restrict Wq,
    const float* __restrict Wkv, const float* __restrict Wp,
    _Float16* __restrict xh, _Float16* __restrict wqh,
    _Float16* __restrict wkh, _Float16* __restrict wvh,
    _Float16* __restrict wph) {
  const int i = blockIdx.x * 256 + threadIdx.x;
  if (i < Bq * Nn * Cc) xh[i] = (_Float16)x[i];
  if (i < Cc * Cc) {
    wqh[i] = (_Float16)Wq[i];
    wph[i] = (_Float16)Wp[i];
    wkh[i] = (_Float16)Wkv[i];             // Wkv rows [0, C)   -> K
    wvh[i] = (_Float16)Wkv[Cc * Cc + i];   // Wkv rows [C, 2C)  -> V
  }
}

// ---------------------------------------------------------------------------
// 2) QKV projection GEMM: [8192 x 128] @ W^T, scatter into per-head layouts.
//    Weight (32KB) staged once into LDS via async copy; A-fragments preloaded;
//    B-fragments software-pipelined one ahead so DS waits hide under WMMA.
//    blockIdx.y: 0=Q, 1=K (both [B,H,N,HD]), 2=V -> Vt [B,H,HD,N]
// ---------------------------------------------------------------------------
__global__ __launch_bounds__(256) void qkv_kernel(
    const _Float16* __restrict xh, const _Float16* __restrict wqh,
    const _Float16* __restrict wkh, const _Float16* __restrict wvh,
    _Float16* __restrict Qb, _Float16* __restrict Kb, _Float16* __restrict Vt) {
  __shared__ __align__(16) _Float16 Wl[Cc * Cc];  // 32 KB
  const int tid  = threadIdx.x;
  const int wave = tid >> 5;
  const int lane = tid & 31;
  const int half = lane >> 4;
  const int col  = lane & 15;
  const int m0   = blockIdx.x * 128 + wave * 16;  // token-row base of this wave
  const int sel  = blockIdx.y;
  const _Float16* W = (sel == 0) ? wqh : ((sel == 1) ? wkh : wvh);

  // stage the whole 128x128 f16 weight into LDS (2048 x 16B chunks)
#pragma unroll
  for (int j = 0; j < 8; ++j) {
    const int chunk = j * 256 + tid;
    async_ld_b128(&Wl[chunk * 8], W + chunk * 8);
  }

  // preload all four A-fragments (this wave's 16x128 strip of x)
  v16h a[4];
#pragma unroll
  for (int kc = 0; kc < 4; ++kc)
    a[kc] = load_fragA(xh + m0 * Cc + kc * 32, Cc);

  v8f acc[8];
#pragma unroll
  for (int t = 0; t < 8; ++t) acc[t] = (v8f){};

  wait_async0();
  __syncthreads();

  // pipelined MAC loop: flatten (kc,t), prefetch B-frag i+1 before WMMA i
  v16h b = load_fragB(Wl, Cc);
#pragma unroll
  for (int i = 0; i < 32; ++i) {
    const int kc = i >> 3;
    const int t  = i & 7;
    v16h bn = b;
    if (i < 31) {
      const int i2 = i + 1;
      bn = load_fragB(Wl + ((i2 & 7) * 16) * Cc + (i2 >> 3) * 32, Cc);
    }
    acc[t] = WMMA16(a[kc], b, acc[t]);
    b = bn;
  }

  const int bidx = m0 >> 11;        // batch
  const int n0   = m0 & (Nn - 1);   // token within batch
#pragma unroll
  for (int t = 0; t < 8; ++t) {
    const int co = t * 16 + col;
    const int h  = co >> 5;
    const int hd = co & 31;
    if (sel == 2) {
      // Vt[((b*H+h)*HD + hd)*N + n]; rows r are contiguous tokens -> b128 store
      _Float16* dst = Vt + ((bidx * Hh + h) * HDd + hd) * Nn + n0 + half * 8;
      v8h vv;
#pragma unroll
      for (int r = 0; r < 8; ++r) vv[r] = (_Float16)acc[t][r];
      *(v8h*)dst = vv;
    } else {
      _Float16* dst =
          (sel == 0 ? Qb : Kb) + ((bidx * Hh + h) * Nn + n0 + half * 8) * HDd + hd;
#pragma unroll
      for (int r = 0; r < 8; ++r) dst[r * HDd] = (_Float16)acc[t][r];
    }
  }
}

// ---------------------------------------------------------------------------
// 3) Flash attention with adjacency mask. One block per (b,h,128-query tile),
//    8 waves x 16 query rows. K / V^T tiles double-buffered in LDS via
//    async global->LDS copies; online softmax; P staged per-wave in LDS.
//    Fragment reads pipelined one ahead of each WMMA.
// ---------------------------------------------------------------------------
__global__ __launch_bounds__(256) void attn_kernel(
    const _Float16* __restrict Qb, const _Float16* __restrict Kb,
    const _Float16* __restrict Vt, const int* __restrict A,
    _Float16* __restrict Ao) {
  __shared__ __align__(16) _Float16 Kt[2][128 * HDd];   // 2 x 8 KB
  __shared__ __align__(16) _Float16 Vl[2][HDd * 128];   // 2 x 8 KB
  __shared__ __align__(16) _Float16 Plds[8 * 16 * 128]; // 32 KB

  const int tid  = threadIdx.x;
  const int wave = tid >> 5;
  const int lane = tid & 31;
  const int half = lane >> 4;
  const int col  = lane & 15;

  constexpr int QBLK = Nn / 128;  // 16
  constexpr int NB   = Nn / 128;  // 16 KV blocks
  const int bh = blockIdx.x / QBLK;
  const int q0 = (blockIdx.x % QBLK) * 128 + wave * 16;

  // cooperative async stage of KV block kb into buffer buf
  auto stage = [&](int kb, int buf) {
    const int k0 = kb * 128;
    // K tile: 128 rows x 32 halves, fully contiguous 8KB -> 512 x 16B chunks
    const _Float16* kg = Kb + (bh * Nn + k0) * HDd;
#pragma unroll
    for (int j = 0; j < 2; ++j) {
      const int chunk = tid * 2 + j;  // 0..511
      async_ld_b128(&Kt[buf][chunk * 8], kg + chunk * 8);
    }
    // V^T tile: 32 rows (d) x 128 halves, row stride Nn -> 512 x 16B chunks
#pragma unroll
    for (int j = 0; j < 2; ++j) {
      const int chunk = tid * 2 + j;  // 0..511
      const int d  = chunk >> 4;      // 0..31
      const int cc = chunk & 15;      // 0..15
      const _Float16* vg = Vt + (bh * HDd + d) * Nn + k0 + cc * 8;
      async_ld_b128(&Vl[buf][chunk * 8], vg);
    }
  };

  const v16h qf = load_fragA(Qb + (bh * Nn + q0) * HDd, HDd);

  v8f o0 = (v8f){}, o1 = (v8f){};
  float mrow[8], lrow[8];
#pragma unroll
  for (int r = 0; r < 8; ++r) { mrow[r] = -__builtin_inff(); lrow[r] = 0.f; }

  _Float16* Pw = Plds + wave * (16 * 128);

  stage(0, 0);

  for (int kb = 0; kb < NB; ++kb) {
    const int cur = kb & 1;
    const int k0  = kb * 128;

    wait_async0();     // my slice of buffer `cur` has landed
    __syncthreads();   // everyone's slice landed; prev buffer fully consumed
    if (kb + 1 < NB) stage(kb + 1, cur ^ 1);  // prefetch next block

    const _Float16* Kl = Kt[cur];
    const _Float16* Vc = Vl[cur];

    // ---- S = Q K^T (8 x v_wmma; K-frags pipelined one ahead)
    v8f s[8];
    v16h kf = load_fragB(Kl, HDd);
#pragma unroll
    for (int t = 0; t < 8; ++t) {
      v16h kn = kf;
      if (t < 7) kn = load_fragB(Kl + (t + 1) * 16 * HDd, HDd);
      v8f z = (v8f){};
      s[t] = WMMA16(qf, kf, z);
      kf = kn;
    }

    // ---- mask, scale, online softmax (row reductions inside 16-lane halves)
#pragma unroll
    for (int r = 0; r < 8; ++r) {
      const int qg = q0 + half * 8 + r;
      const int* ar = A + qg * Nn + k0 + col;
      float vals[8];
      float mx = -__builtin_inff();
#pragma unroll
      for (int t = 0; t < 8; ++t) {
        float v = s[t][r] * SCALE;
        if (ar[t * 16] <= 0) v = NEGV;
        vals[t] = v;
        mx = fmaxf(mx, v);
      }
#pragma unroll
      for (int off = 1; off < 16; off <<= 1)
        mx = fmaxf(mx, __shfl_xor(mx, off, 32));
      const float mnew = fmaxf(mrow[r], mx);
      const float corr = __expf(mrow[r] - mnew);
      mrow[r] = mnew;
      float ls = 0.f;
#pragma unroll
      for (int t = 0; t < 8; ++t) {
        const float p = __expf(vals[t] - mnew);
        ls += p;
        Pw[(half * 8 + r) * 128 + t * 16 + col] = (_Float16)p;
      }
#pragma unroll
      for (int off = 1; off < 16; off <<= 1) ls += __shfl_xor(ls, off, 32);
      lrow[r] = lrow[r] * corr + ls;
      o0[r] *= corr;
      o1[r] *= corr;
    }

    // ---- O += P @ V (P + V^T both from LDS; frags pipelined one ahead)
    v16h pf = load_fragA(Pw, 128);
    v16h v0 = load_fragB(Vc, 128);
    v16h v1 = load_fragB(Vc + 16 * 128, 128);
#pragma unroll
    for (int kc = 0; kc < 4; ++kc) {
      v16h pfn = pf, v0n = v0, v1n = v1;
      if (kc < 3) {
        pfn = load_fragA(Pw + (kc + 1) * 32, 128);
        v0n = load_fragB(Vc + (kc + 1) * 32, 128);
        v1n = load_fragB(Vc + 16 * 128 + (kc + 1) * 32, 128);
      }
      o0 = WMMA16(pf, v0, o0);
      o1 = WMMA16(pf, v1, o1);
      pf = pfn; v0 = v0n; v1 = v1n;
    }
  }

  // ---- normalize and store attention output [B,H,N,HD]
#pragma unroll
  for (int r = 0; r < 8; ++r) {
    const float inv = 1.0f / lrow[r];
    const int tok = q0 + half * 8 + r;
    _Float16* dst = Ao + (bh * Nn + tok) * HDd;
    dst[col]      = (_Float16)(o0[r] * inv);
    dst[16 + col] = (_Float16)(o1[r] * inv);
  }
}

// ---------------------------------------------------------------------------
// 4) Output projection + bias + residual (fp32 epilogue to d_out).
//    Wp staged in LDS via async copy, A-fragments preloaded, B pipelined.
// ---------------------------------------------------------------------------
__global__ __launch_bounds__(256) void proj_kernel(
    const _Float16* __restrict Ao, const _Float16* __restrict wph,
    const float* __restrict bp, const float* __restrict x,
    float* __restrict out) {
  __shared__ __align__(16) _Float16 Wl[Cc * Cc];  // 32 KB
  const int tid  = threadIdx.x;
  const int wave = tid >> 5;
  const int lane = tid & 31;
  const int half = lane >> 4;
  const int col  = lane & 15;
  const int m0   = blockIdx.x * 128 + wave * 16;
  const int bidx = m0 >> 11;
  const int n0   = m0 & (Nn - 1);

#pragma unroll
  for (int j = 0; j < 8; ++j) {
    const int chunk = j * 256 + tid;
    async_ld_b128(&Wl[chunk * 8], wph + chunk * 8);
  }

  v16h a[4];
#pragma unroll
  for (int kc = 0; kc < 4; ++kc)  // K-chunk == head (c = kc*32 + hd)
    a[kc] = load_fragA(Ao + ((bidx * Hh + kc) * Nn + n0) * HDd, HDd);

  v8f acc[8];
#pragma unroll
  for (int t = 0; t < 8; ++t) acc[t] = (v8f){};

  wait_async0();
  __syncthreads();

  v16h b = load_fragB(Wl, Cc);
#pragma unroll
  for (int i = 0; i < 32; ++i) {
    const int kc = i >> 3;
    const int t  = i & 7;
    v16h bn = b;
    if (i < 31) {
      const int i2 = i + 1;
      bn = load_fragB(Wl + ((i2 & 7) * 16) * Cc + (i2 >> 3) * 32, Cc);
    }
    acc[t] = WMMA16(a[kc], b, acc[t]);
    b = bn;
  }

#pragma unroll
  for (int t = 0; t < 8; ++t) {
    const int co = t * 16 + col;
    const float bias = bp[co];
#pragma unroll
    for (int r = 0; r < 8; ++r) {
      const int m = m0 + half * 8 + r;
      out[m * Cc + co] = acc[t][r] + bias + x[m * Cc + co];
    }
  }
}

// ---------------------------------------------------------------------------
// Host-side launcher (graph-capture safe: only kernel launches on `stream`)
// ---------------------------------------------------------------------------
extern "C" void kernel_launch(void* const* d_in, const int* in_sizes, int n_in,
                              void* d_out, int out_size, void* d_ws, size_t ws_size,
                              hipStream_t stream) {
  const float* x   = (const float*)d_in[0];
  const int*   A   = (const int*)d_in[1];
  const float* Wq  = (const float*)d_in[2];
  const float* Wkv = (const float*)d_in[3];
  const float* Wp  = (const float*)d_in[4];
  const float* bp  = (const float*)d_in[5];
  float* out = (float*)d_out;

  char* ws = (char*)d_ws;
  _Float16* xh  = (_Float16*)(ws + 0x000000);  // 2 MB
  _Float16* wqh = (_Float16*)(ws + 0x200000);  // 32 KB
  _Float16* wkh = (_Float16*)(ws + 0x208000);
  _Float16* wvh = (_Float16*)(ws + 0x210000);
  _Float16* wph = (_Float16*)(ws + 0x218000);
  _Float16* Qb  = (_Float16*)(ws + 0x220000);  // 2 MB  [B,H,N,HD]
  _Float16* Kb  = (_Float16*)(ws + 0x420000);  // 2 MB  [B,H,N,HD]
  _Float16* Vt  = (_Float16*)(ws + 0x620000);  // 2 MB  [B,H,HD,N]
  _Float16* Ao  = (_Float16*)(ws + 0x820000);  // 2 MB  [B,H,N,HD]

  cvt_kernel<<<(Bq * Nn * Cc + 255) / 256, 256, 0, stream>>>(
      x, Wq, Wkv, Wp, xh, wqh, wkh, wvh, wph);
  qkv_kernel<<<dim3((Bq * Nn) / 128, 3), 256, 0, stream>>>(
      xh, wqh, wkh, wvh, Qb, Kb, Vt);
  attn_kernel<<<Bq * Hh * (Nn / 128), 256, 0, stream>>>(Qb, Kb, Vt, A, Ao);
  proj_kernel<<<(Bq * Nn) / 128, 256, 0, stream>>>(Ao, wph, bp, x, out);
}
